// GCN_77730318123074
// MI455X (gfx1250) — matrix operand order repaired and checked by reading, not compiled, not run on hardware
//
#include <hip/hip_runtime.h>
#include <hip/hip_bf16.h>

// ---------------------------------------------------------------------------
// Types for CDNA5 WMMA (wave32): 16x16x32 bf16 -> f32
// ---------------------------------------------------------------------------
typedef __attribute__((ext_vector_type(16))) __bf16 v16bf;
typedef __attribute__((ext_vector_type(8)))  float  v8f;

union BfVec {
    v16bf v;
    uint4 q[2];
};

#define WAVES_PER_BLOCK 4

#if __has_builtin(__builtin_amdgcn_global_load_async_to_lds_b128)
#define HAVE_ASYNC_LDS 1
// Signature (from clang diagnostic): (int4 AS(1)*, int4 AS(3)*, imm int, imm int)
typedef int v4i_a __attribute__((ext_vector_type(4)));
typedef __attribute__((address_space(1))) v4i_a* g_v4i_ptr;
typedef __attribute__((address_space(3))) v4i_a* l_v4i_ptr;
#endif

// ---------------------------------------------------------------------------
// Utility kernels
// ---------------------------------------------------------------------------
__global__ __launch_bounds__(256) void zero_f32(float* __restrict__ p, long long n) {
    long long i = (long long)blockIdx.x * blockDim.x + threadIdx.x;
    long long stride = (long long)gridDim.x * blockDim.x;
    for (; i < n; i += stride) p[i] = 0.0f;
}

__global__ __launch_bounds__(256) void cast_f32_bf16(const float* __restrict__ in,
                                                     __bf16* __restrict__ out, long long n) {
    long long i = (long long)blockIdx.x * blockDim.x + threadIdx.x;
    long long stride = (long long)gridDim.x * blockDim.x;
    for (; i < n; i += stride) out[i] = (__bf16)in[i];
}

// W: [K, C] row-major fp32  ->  WT: [C, K] row-major bf16 (so WMMA B-operand
// loads are K-contiguous per output column).
__global__ __launch_bounds__(256) void transpose_cast(const float* __restrict__ W,
                                                      __bf16* __restrict__ WT,
                                                      int K, int C) {
    int i = blockIdx.x * blockDim.x + threadIdx.x;
    if (i >= K * C) return;
    int k = i / C;
    int c = i - k * C;
    WT[(size_t)c * K + k] = (__bf16)W[i];
}

// ---------------------------------------------------------------------------
// Edge aggregation: agg[dst] += x[src], fp32 atomics (fits in 192MB L2).
// One thread per edge; indices read once, channels looped in-register.
// ---------------------------------------------------------------------------
template<int C>
__global__ __launch_bounds__(256) void scatter_add(const float* __restrict__ x,
                                                   const int* __restrict__ src,
                                                   const int* __restrict__ dst,
                                                   float* __restrict__ agg, int E) {
    int e = blockIdx.x * blockDim.x + threadIdx.x;
    if (e >= E) return;
    const int s = src[e];
    const int d = dst[e];
    const float* xi = x + (size_t)s * C;
    float* o = agg + (size_t)d * C;
#pragma unroll 4
    for (int c = 0; c < C; c += 4) {
        float4 v = *(const float4*)(xi + c);
        atomicAdd(o + c + 0, v.x);
        atomicAdd(o + c + 1, v.y);
        atomicAdd(o + c + 2, v.z);
        atomicAdd(o + c + 3, v.w);
    }
}

// ---------------------------------------------------------------------------
// Fused dual GEMM + bias (+ReLU):  D = A1*B1 + A2*B2 + bias
//   A1,A2:   [M, K] bf16 row-major (agg and x)
//   B1T,B2T: [C, K] bf16 (pre-transposed weights) -- staged into LDS once per
//            block via gfx1250 async global->LDS DMA (ASYNCcnt).
// Each wave owns one 16-row M tile; its A fragments for the full K dimension
// are register-resident and reused across all C/16 N tiles, whose B fragments
// come from LDS (ds_load_b128). K loop fully unrolled (template).
// VGPR layouts per CDNA5 ISA 7.12.2 (wave32):
//   A: lane L holds row M = m0 + (L&15); elems 0..7 -> K = kb+(L>>4)*8+e,
//      elems 8..15 -> K = kb+16+(L>>4)*8+(e-8)   (two 16B loads)
//   B: lane L holds col N = n0 + (L&15); same K mapping as A (two 16B loads)
//   C/D f32: VGPR r -> row m0 + r + 8*(L>>4), col n0 + (L&15)
// ---------------------------------------------------------------------------
template<int K, int C, bool RELU, bool WRITE_BF16>
__global__ __launch_bounds__(WAVES_PER_BLOCK * 32) void gemm_dual_lds(
    const __bf16* __restrict__ A1, const __bf16* __restrict__ B1T,
    const __bf16* __restrict__ A2, const __bf16* __restrict__ B2T,
    const float* __restrict__ bias,
    float* __restrict__ outF, __bf16* __restrict__ outB, int Mtiles) {
    constexpr int KSTEPS = K / 32;
    constexpr int NT = C / 16;
    __shared__ alignas(16) __bf16 sB[2 * C * K];   // [B1T | B2T], each [C][K]

    // ---- stage both weight matrices into LDS (once per block) ----
    {
        const int vecs = (C * K) / 8;      // 16B (8 x bf16) per transfer
#if defined(HAVE_ASYNC_LDS)
        for (int i = threadIdx.x; i < vecs; i += blockDim.x) {
            __builtin_amdgcn_global_load_async_to_lds_b128(
                (g_v4i_ptr)(B1T + (size_t)i * 8),
                (l_v4i_ptr)(sB + (size_t)i * 8), 0, 0);
            __builtin_amdgcn_global_load_async_to_lds_b128(
                (g_v4i_ptr)(B2T + (size_t)i * 8),
                (l_v4i_ptr)(sB + (size_t)(C * K) + (size_t)i * 8), 0, 0);
        }
        asm volatile("s_wait_asynccnt 0x0" ::: "memory");
#else
        for (int i = threadIdx.x; i < vecs; i += blockDim.x) {
            ((uint4*)sB)[i]        = ((const uint4*)B1T)[i];
            ((uint4*)sB)[vecs + i] = ((const uint4*)B2T)[i];
        }
#endif
    }
    __syncthreads();

    const int wave = threadIdx.x >> 5;
    const int lane = threadIdx.x & 31;
    const int mt = blockIdx.x * WAVES_PER_BLOCK + wave;
    if (mt < Mtiles) {                     // wave-uniform: EXEC all-1s inside
        const int m0 = mt << 4;
        const int h  = lane >> 4;
        const int ln = lane & 15;

        // A fragments for the full K dimension: loaded once, kept in VGPRs.
        BfVec a1[KSTEPS], a2[KSTEPS];
        const __bf16* pa1 = A1 + (size_t)(m0 + ln) * K + h * 8;
        const __bf16* pa2 = A2 + (size_t)(m0 + ln) * K + h * 8;
#pragma unroll
        for (int s = 0; s < KSTEPS; ++s) {
            a1[s].q[0] = *(const uint4*)(pa1 + s * 32);
            a1[s].q[1] = *(const uint4*)(pa1 + s * 32 + 16);
            a2[s].q[0] = *(const uint4*)(pa2 + s * 32);
            a2[s].q[1] = *(const uint4*)(pa2 + s * 32 + 16);
        }

        const __bf16* sB1 = sB;
        const __bf16* sB2 = sB + C * K;
        for (int nt = 0; nt < NT; ++nt) {
            const int n0 = nt << 4;
            const __bf16* b1 = sB1 + (size_t)(n0 + ln) * K + h * 8;
            const __bf16* b2 = sB2 + (size_t)(n0 + ln) * K + h * 8;
            v8f acc = {};
#pragma unroll
            for (int s = 0; s < KSTEPS; ++s) {
                BfVec bv;
                bv.q[0] = *(const uint4*)(b1 + s * 32);
                bv.q[1] = *(const uint4*)(b1 + s * 32 + 16);
                acc = __builtin_amdgcn_wmma_f32_16x16x32_bf16(false, a1[s].v, false, bv.v,
                                                              (short)0, acc, false, false);
                bv.q[0] = *(const uint4*)(b2 + s * 32);
                bv.q[1] = *(const uint4*)(b2 + s * 32 + 16);
                acc = __builtin_amdgcn_wmma_f32_16x16x32_bf16(false, a2[s].v, false, bv.v,
                                                              (short)0, acc, false, false);
            }
            const int col = n0 + ln;
            const float bb = bias[col];
#pragma unroll
            for (int r = 0; r < 8; ++r) {
                float v = acc[r] + bb;
                if (RELU) v = v > 0.0f ? v : 0.0f;
                const size_t off = (size_t)(m0 + r + 8 * h) * C + col;
                outF[off] = v;
                if (WRITE_BF16) outB[off] = (__bf16)v;
            }
        }
    }
}

// ---------------------------------------------------------------------------
// Global mean pool + head
// ---------------------------------------------------------------------------
__global__ __launch_bounds__(256) void pool_sum(const float* __restrict__ hh,
                                                const int* __restrict__ batch,
                                                float* __restrict__ psum, int Nn) {
    int i = blockIdx.x * blockDim.x + threadIdx.x;
    if (i >= Nn) return;
    const int g = batch[i];
    const float* hi = hh + (size_t)i * 128;
    float* o = psum + (size_t)g * 128;
#pragma unroll 4
    for (int c = 0; c < 128; c += 4) {
        float4 v = *(const float4*)(hi + c);
        atomicAdd(o + c + 0, v.x);
        atomicAdd(o + c + 1, v.y);
        atomicAdd(o + c + 2, v.z);
        atomicAdd(o + c + 3, v.w);
    }
}

__global__ __launch_bounds__(256) void pool_cnt(const int* __restrict__ batch,
                                                float* __restrict__ cnt, int Nn) {
    int i = blockIdx.x * blockDim.x + threadIdx.x;
    if (i < Nn) atomicAdd(cnt + batch[i], 1.0f);
}

// t1[g, j] = (psum[g,:]/cnt[g]) @ W1[:, j] + b1[j]   (K=128, 40 outputs)
__global__ __launch_bounds__(64) void head1(const float* __restrict__ psum,
                                            const float* __restrict__ cnt,
                                            const float* __restrict__ W1,
                                            const float* __restrict__ b1,
                                            float* __restrict__ t1) {
    int g = blockIdx.x;
    int j = threadIdx.x;
    if (j >= 40) return;
    float inv = 1.0f / fmaxf(cnt[g], 1.0f);
    float s = b1[j];
    const float* pg = psum + (size_t)g * 128;
    for (int k = 0; k < 128; ++k) s += pg[k] * inv * W1[k * 40 + j];
    t1[g * 40 + j] = s;
}

// out[g, j] = t1[g,:] @ W2[:, j] + b2[j]   (K=40, 10 outputs)
__global__ __launch_bounds__(32) void head2(const float* __restrict__ t1,
                                            const float* __restrict__ W2,
                                            const float* __restrict__ b2,
                                            float* __restrict__ out) {
    int g = blockIdx.x;
    int j = threadIdx.x;
    if (j >= 10) return;
    float s = b2[j];
    const float* tg = t1 + (size_t)g * 40;
    for (int k = 0; k < 40; ++k) s += tg[k] * W2[k * 10 + j];
    out[g * 10 + j] = s;
}

// ---------------------------------------------------------------------------
// Host orchestration
// ---------------------------------------------------------------------------
extern "C" void kernel_launch(void* const* d_in, const int* in_sizes, int n_in,
                              void* d_out, int out_size, void* d_ws, size_t ws_size,
                              hipStream_t stream) {
    const float* x    = (const float*)d_in[0];
    const int*   ei   = (const int*)d_in[1];
    const int*   bat  = (const int*)d_in[2];
    const float* Wr1  = (const float*)d_in[3];
    const float* br1  = (const float*)d_in[4];
    const float* Wo1  = (const float*)d_in[5];
    const float* Wr2  = (const float*)d_in[6];
    const float* br2  = (const float*)d_in[7];
    const float* Wo2  = (const float*)d_in[8];
    const float* Wr3  = (const float*)d_in[9];
    const float* br3  = (const float*)d_in[10];
    const float* Wo3  = (const float*)d_in[11];
    const float* W1   = (const float*)d_in[12];
    const float* b1   = (const float*)d_in[13];
    const float* W2   = (const float*)d_in[14];
    const float* b2   = (const float*)d_in[15];
    float* out = (float*)d_out;

    const int Nn = in_sizes[0] / 128;   // 50000 nodes
    const int Ee = in_sizes[1] / 2;     // 1600000 edges
    const int Gg = 512;                 // graphs
    const int* src = ei;
    const int* dst = ei + Ee;

    // Workspace carve-up (256B aligned regions)
    char* ws = (char*)d_ws;
    size_t off = 0;
    auto carve = [&](size_t bytes) -> char* {
        char* p = ws + off;
        off += (bytes + 255) & ~(size_t)255;
        return p;
    };
    __bf16* xb   = (__bf16*)carve((size_t)Nn * 128 * 2);
    float*  h1f  = (float*) carve((size_t)Nn * 128 * 4);
    __bf16* h1b  = (__bf16*)carve((size_t)Nn * 128 * 2);
    float*  h2f  = (float*) carve((size_t)Nn * 256 * 4);
    __bf16* h2b  = (__bf16*)carve((size_t)Nn * 256 * 2);
    float*  h3f  = (float*) carve((size_t)Nn * 128 * 4);
    float*  aggf = (float*) carve((size_t)Nn * 256 * 4);
    __bf16* aggb = (__bf16*)carve((size_t)Nn * 256 * 2);
    __bf16* Wr1T = (__bf16*)carve((size_t)128 * 128 * 2);
    __bf16* Wo1T = (__bf16*)carve((size_t)128 * 128 * 2);
    __bf16* Wr2T = (__bf16*)carve((size_t)128 * 256 * 2);
    __bf16* Wo2T = (__bf16*)carve((size_t)128 * 256 * 2);
    __bf16* Wr3T = (__bf16*)carve((size_t)256 * 128 * 2);
    __bf16* Wo3T = (__bf16*)carve((size_t)256 * 128 * 2);
    float*  psum = (float*) carve((size_t)Gg * 128 * 4);
    float*  pcnt = (float*) carve((size_t)Gg * 4);
    float*  t1   = (float*) carve((size_t)Gg * 40 * 4);
    (void)ws_size; (void)n_in; (void)out_size;

    const int Mtiles  = Nn / 16;                       // 3125
    const int gemm_gx = (Mtiles + WAVES_PER_BLOCK - 1) / WAVES_PER_BLOCK;
    const int eblocks = (Ee + 255) / 256;
    const int nblocks = (Nn + 255) / 256;

    // -- prep: bf16 casts + transposed bf16 weights
    cast_f32_bf16<<<2048, 256, 0, stream>>>(x, xb, (long long)Nn * 128);
    transpose_cast<<<(128 * 128 + 255) / 256, 256, 0, stream>>>(Wr1, Wr1T, 128, 128);
    transpose_cast<<<(128 * 128 + 255) / 256, 256, 0, stream>>>(Wo1, Wo1T, 128, 128);
    transpose_cast<<<(128 * 256 + 255) / 256, 256, 0, stream>>>(Wr2, Wr2T, 128, 256);
    transpose_cast<<<(128 * 256 + 255) / 256, 256, 0, stream>>>(Wo2, Wo2T, 128, 256);
    transpose_cast<<<(256 * 128 + 255) / 256, 256, 0, stream>>>(Wr3, Wr3T, 256, 128);
    transpose_cast<<<(256 * 128 + 255) / 256, 256, 0, stream>>>(Wo3, Wo3T, 256, 128);

    // -- layer 1: agg = scatter(x); h1 = relu(agg@Wr1 + x@Wo1 + br1)
    zero_f32<<<2048, 256, 0, stream>>>(aggf, (long long)Nn * 128);
    scatter_add<128><<<eblocks, 256, 0, stream>>>(x, src, dst, aggf, Ee);
    cast_f32_bf16<<<2048, 256, 0, stream>>>(aggf, aggb, (long long)Nn * 128);
    gemm_dual_lds<128, 128, true, true><<<gemm_gx, 128, 0, stream>>>(
        aggb, Wr1T, xb, Wo1T, br1, h1f, h1b, Mtiles);

    // -- layer 2: agg = scatter(h1); h2 = relu(agg@Wr2 + h1@Wo2 + br2)
    zero_f32<<<2048, 256, 0, stream>>>(aggf, (long long)Nn * 128);
    scatter_add<128><<<eblocks, 256, 0, stream>>>(h1f, src, dst, aggf, Ee);
    cast_f32_bf16<<<2048, 256, 0, stream>>>(aggf, aggb, (long long)Nn * 128);
    gemm_dual_lds<128, 256, true, true><<<gemm_gx, 128, 0, stream>>>(
        aggb, Wr2T, h1b, Wo2T, br2, h2f, h2b, Mtiles);

    // -- layer 3: agg = scatter(h2); h3 = agg@Wr3 + h2@Wo3 + br3 (no relu)
    zero_f32<<<2048, 256, 0, stream>>>(aggf, (long long)Nn * 256);
    scatter_add<256><<<eblocks, 256, 0, stream>>>(h2f, src, dst, aggf, Ee);
    cast_f32_bf16<<<2048, 256, 0, stream>>>(aggf, aggb, (long long)Nn * 256);
    gemm_dual_lds<256, 128, false, false><<<gemm_gx, 128, 0, stream>>>(
        aggb, Wr3T, h2b, Wo3T, br3, h3f, nullptr, Mtiles);

    // -- global mean pool (psum and pcnt are adjacent: zero both in one pass)
    zero_f32<<<512, 256, 0, stream>>>(psum, (long long)Gg * 128 + Gg);
    pool_sum<<<nblocks, 256, 0, stream>>>(h3f, bat, psum, Nn);
    pool_cnt<<<nblocks, 256, 0, stream>>>(bat, pcnt, Nn);

    // -- head
    head1<<<Gg, 64, 0, stream>>>(psum, pcnt, W1, b1, t1);
    head2<<<Gg, 32, 0, stream>>>(t1, W2, b2, out);
}
